// Attention_42176578846883
// MI455X (gfx1250) — matrix operand order repaired and checked by reading, not compiled, not run on hardware
//
#include <hip/hip_runtime.h>

typedef __bf16 bf16;
typedef __attribute__((ext_vector_type(16))) __bf16 v16bf;
typedef __attribute__((ext_vector_type(8)))  __bf16 v8bf;
typedef __attribute__((ext_vector_type(8)))  float  v8f;
typedef __attribute__((ext_vector_type(4)))  float  v4f;

#define DIMX   1024
#define HEADS  16
#define DHEAD  64
#define SEQ    2048
#define BATCH  2
#define ROWS   (BATCH * SEQ)      // 4096
#define NQKV   (3 * DIMX)         // 3072
#define SCLOG2 0.180336884f       // DHEAD^-0.5 * log2(e): softmax done in exp2 domain

__device__ __forceinline__ bf16 f2bf(float f) {
    unsigned u = __builtin_bit_cast(unsigned, f);
    u += 0x7FFFu + ((u >> 16) & 1u);           // round-to-nearest-even
    unsigned short s = (unsigned short)(u >> 16);
    return __builtin_bit_cast(bf16, s);
}

// Async 16B copy global -> LDS (CDNA5, tracked by ASYNCcnt).
// Low 32 bits of a generic pointer to __shared__ are the LDS byte offset.
__device__ __forceinline__ void async_ld_b128(const bf16* lds_dst, const bf16* gsrc) {
    unsigned loff = (unsigned)(uintptr_t)lds_dst;
    unsigned long long g = (unsigned long long)(uintptr_t)gsrc;
    asm volatile("global_load_async_to_lds_b128 %0, %1, off"
                 :: "v"(loff), "v"(g) : "memory");
}

// A fragment: 16x32 bf16 (M x K). lane = M (mod 16); regs 0..7 -> K = half*8+i,
// regs 8..15 -> K = 16+half*8+i  (half = lane>>4).  Two 16B LDS loads.
__device__ __forceinline__ v16bf load_a_frag(const bf16* base, int ld) {
    int lane = threadIdx.x & 31;
    int row  = lane & 15;
    int half = lane >> 4;
    const bf16* p = base + row * ld + half * 8;
    v8bf lo = *(const v8bf*)p;
    v8bf hi = *(const v8bf*)(p + 16);
    v16bf f;
#pragma unroll
    for (int i = 0; i < 8; ++i) { f[i] = lo[i]; f[i + 8] = hi[i]; }
    return f;
}

// B fragment: 32x16 bf16 (K x N). lane = N (mod 16); regs 0..7 -> K = half*8+i,
// regs 8..15 -> K = 16+half*8+i  (half = lane>>4).
__device__ __forceinline__ v16bf load_b_frag(const bf16* base, int ld) {
    int lane = threadIdx.x & 31;
    const bf16* p = base + (lane & 15) + (lane >> 4) * 8 * ld;
    v16bf f;
#pragma unroll
    for (int i = 0; i < 8; ++i) { f[i] = p[i * ld]; f[i + 8] = p[(16 + i) * ld]; }
    return f;
}

// ---------------------------------------------------------------- f32 -> bf16
__global__ __launch_bounds__(256)
void cvt_bf16_kernel(const float* __restrict__ src, bf16* __restrict__ dst, int n) {
    for (int i = blockIdx.x * 256 + threadIdx.x; i < n; i += gridDim.x * 256)
        dst[i] = f2bf(src[i]);
}

// --------------------------------------------------------- LayerNorm -> bf16
__global__ __launch_bounds__(256)
void ln_kernel(const float* __restrict__ x, const float* __restrict__ gamma,
               const float* __restrict__ beta, bf16* __restrict__ xn) {
    __shared__ float s1[256], s2[256];
    const int row = blockIdx.x, tid = threadIdx.x;
    const v4f* x4 = (const v4f*)(x + (size_t)row * DIMX);
    v4f v = x4[tid];
    s1[tid] = v[0] + v[1] + v[2] + v[3];
    s2[tid] = v[0]*v[0] + v[1]*v[1] + v[2]*v[2] + v[3]*v[3];
    __syncthreads();
#pragma unroll
    for (int st = 128; st > 0; st >>= 1) {
        if (tid < st) { s1[tid] += s1[tid + st]; s2[tid] += s2[tid + st]; }
        __syncthreads();
    }
    float mu   = s1[0] * (1.0f / DIMX);
    float var  = s2[0] * (1.0f / DIMX) - mu * mu;
    float rstd = rsqrtf(var + 1e-5f);
    const v4f* g4 = (const v4f*)gamma;
    const v4f* b4 = (const v4f*)beta;
    v4f g = g4[tid], bb = b4[tid];
#pragma unroll
    for (int j = 0; j < 4; ++j)
        xn[(size_t)row * DIMX + tid * 4 + j] = f2bf((v[j] - mu) * rstd * g[j] + bb[j]);
}

// ------------------------------------------------------ tiled WMMA bf16 GEMM
// 128x128 workgroup tile, 8 waves (4 in M x 2 in N), 32x64 per wave.
// Double-buffered LDS tiles staged with async-to-LDS copies.
// mode 0: scatter bf16 into qkv [3][b,h,n,64];  mode 1: f32 out + bias.
__global__ __launch_bounds__(256)
void gemm_bf16(const bf16* __restrict__ A, const bf16* __restrict__ B,
               int lda, int ldb, int K, int mode,
               bf16* __restrict__ qkvOut, float* __restrict__ fOut,
               const float* __restrict__ bias, int Nc) {
    __shared__ __align__(16) bf16 As[2][128 * 40];   // stride 40: conflict-free A frags
    __shared__ __align__(16) bf16 Bs[2][32 * 136];   // stride 136: conflict-free B frags
    const int tid    = threadIdx.x;
    const int blockM = blockIdx.y * 128;
    const int blockN = blockIdx.x * 128;
    const int wave = tid >> 5;
    const int wm = wave & 3, wn = wave >> 2;
    const int lane = tid & 31, half = lane >> 4, ln = lane & 15;

    // per-thread staging assignment (straight-line, no guards)
    const int ar = tid >> 2, ac = (tid & 3) * 8;     // A: rows ar, ar+64; 8-col chunk
    const int br = tid >> 4, bc = (tid & 15) * 8;    // B: rows br, br+16; 8-col chunk
    const bf16* Arow0 = A + (size_t)(blockM + ar) * lda + ac;
    const bf16* Arow1 = A + (size_t)(blockM + ar + 64) * lda + ac;
    const bf16* Brow0 = B + (size_t)br * ldb + blockN + bc;
    const bf16* Brow1 = B + (size_t)(br + 16) * ldb + blockN + bc;

    v8f acc[2][4] = {};
    const int kTiles = K >> 5;

    // prefetch tile 0
    {
        async_ld_b128(&As[0][ar * 40 + ac],         Arow0);
        async_ld_b128(&As[0][(ar + 64) * 40 + ac],  Arow1);
        async_ld_b128(&Bs[0][br * 136 + bc],        Brow0);
        async_ld_b128(&Bs[0][(br + 16) * 136 + bc], Brow1);
    }

    for (int kt = 0; kt < kTiles; ++kt) {
        const int buf = kt & 1;
        if (kt + 1 < kTiles) {
            const int ko = (kt + 1) << 5;
            async_ld_b128(&As[buf ^ 1][ar * 40 + ac],         Arow0 + ko);
            async_ld_b128(&As[buf ^ 1][(ar + 64) * 40 + ac],  Arow1 + ko);
            async_ld_b128(&Bs[buf ^ 1][br * 136 + bc],        Brow0 + (size_t)ko * ldb);
            async_ld_b128(&Bs[buf ^ 1][(br + 16) * 136 + bc], Brow1 + (size_t)ko * ldb);
            asm volatile("s_wait_asynccnt 0x4" ::: "memory");  // tile kt done, kt+1 in flight
        } else {
            asm volatile("s_wait_asynccnt 0x0" ::: "memory");
        }
        __syncthreads();

        v16bf af[2];
#pragma unroll
        for (int mi = 0; mi < 2; ++mi)
            af[mi] = load_a_frag(&As[buf][(wm * 32 + mi * 16) * 40], 40);
        v16bf bfr[4];
#pragma unroll
        for (int ni = 0; ni < 4; ++ni)
            bfr[ni] = load_b_frag(&Bs[buf][wn * 64 + ni * 16], 136);
#pragma unroll
        for (int mi = 0; mi < 2; ++mi)
#pragma unroll
            for (int ni = 0; ni < 4; ++ni)
                acc[mi][ni] = __builtin_amdgcn_wmma_f32_16x16x32_bf16(
                    false, af[mi], false, bfr[ni], (short)0, acc[mi][ni], false, false);
        __syncthreads();   // all waves done reading buf before it is overwritten
    }

#pragma unroll
    for (int mi = 0; mi < 2; ++mi)
#pragma unroll
        for (int ni = 0; ni < 4; ++ni)
#pragma unroll
            for (int r = 0; r < 8; ++r) {
                int row = blockM + wm * 32 + mi * 16 + r + 8 * half;
                int col = blockN + wn * 64 + ni * 16 + ln;
                float v = acc[mi][ni][r];
                if (mode == 0) {
                    int sel = col >> 10, rem = col & 1023;
                    int h = rem >> 6, d = rem & 63;
                    int b = row >> 11, n = row & 2047;
                    qkvOut[(size_t)sel * (BATCH * HEADS * SEQ * DHEAD)
                           + ((size_t)(b * HEADS + h) * SEQ + n) * DHEAD + d] = f2bf(v);
                } else {
                    fOut[(size_t)row * Nc + col] = v + bias[col];
                }
            }
}

// --------------------------------------------- flash attention (bf16 WMMA)
// grid: (SEQ/64, BATCH*HEADS); 128 threads = 4 waves, 16 query rows per wave.
// Double-buffered K/V: V(jt+1) streams via async-to-LDS, K(jt+1) prefetched
// into registers and store-transposed after compute.
__global__ __launch_bounds__(128)
void attn_kernel(const bf16* __restrict__ qg, const bf16* __restrict__ kg,
                 const bf16* __restrict__ vg, bf16* __restrict__ sa) {
    __shared__ __align__(16) bf16 Qs[64 * 72];        // queries, natural layout
    __shared__ __align__(16) bf16 Kt[2][64 * 40];     // K block transposed: [dim][key]
    __shared__ __align__(16) bf16 Vs[2][32 * 72];     // V block natural: [key][dim]
    __shared__ __align__(16) bf16 Ps[4][16 * 40];     // per-wave P C->A relayout

    const int tid  = threadIdx.x;
    const int wave = tid >> 5;
    const int lane = tid & 31, half = lane >> 4, ln = lane & 15;
    const int bh = blockIdx.y;
    const int b  = bh >> 4, h = bh & 15;
    const int q0 = blockIdx.x * 64;

    const bf16* Q  = qg + (size_t)bh * SEQ * DHEAD;
    const bf16* Kp = kg + (size_t)bh * SEQ * DHEAD;
    const bf16* Vp = vg + (size_t)bh * SEQ * DHEAD;

    // per-thread K/V staging assignment: 2 chunks of 8 dims
    const int key0 = tid >> 3, c0 = (tid & 7) * 8;          // i = tid
    const int key1 = (tid + 128) >> 3, c1 = c0;             // i = tid + 128

#pragma unroll
    for (int ii = 0; ii < 4; ++ii) {                  // stage 64x64 Q tile (async)
        int i = tid + ii * 128;
        int r = i >> 3, c = (i & 7) * 8;
        async_ld_b128(&Qs[r * 72 + c], &Q[(size_t)(q0 + r) * DHEAD + c]);
    }
    // stage V(0) (async) + K(0) (regs -> LDS transposed)
    async_ld_b128(&Vs[0][key0 * 72 + c0], &Vp[(size_t)key0 * DHEAD + c0]);
    async_ld_b128(&Vs[0][key1 * 72 + c1], &Vp[(size_t)key1 * DHEAD + c1]);
    {
        v8bf t0 = *(const v8bf*)&Kp[(size_t)key0 * DHEAD + c0];
        v8bf t1 = *(const v8bf*)&Kp[(size_t)key1 * DHEAD + c1];
#pragma unroll
        for (int j = 0; j < 8; ++j) {
            Kt[0][(c0 + j) * 40 + key0] = t0[j];
            Kt[0][(c1 + j) * 40 + key1] = t1[j];
        }
    }
    asm volatile("s_wait_asynccnt 0x0" ::: "memory");
    __syncthreads();

    v16bf qa[2];
    qa[0] = load_a_frag(&Qs[(wave * 16) * 72 + 0], 72);
    qa[1] = load_a_frag(&Qs[(wave * 16) * 72 + 32], 72);

    v8f o[4] = {};
    float m[8], l[8];
#pragma unroll
    for (int r = 0; r < 8; ++r) { m[r] = -3.0e38f; l[r] = 0.0f; }

    const int jTiles = SEQ / 32;
    for (int jt = 0; jt < jTiles; ++jt) {
        const int buf = jt & 1;
        const bool more = (jt + 1) < jTiles;
        v8bf k0, k1;
        if (more) {                                   // prefetch next K/V tile
            const size_t jo = (size_t)(jt + 1) * 32 * DHEAD;
            async_ld_b128(&Vs[buf ^ 1][key0 * 72 + c0], &Vp[jo + (size_t)key0 * DHEAD + c0]);
            async_ld_b128(&Vs[buf ^ 1][key1 * 72 + c1], &Vp[jo + (size_t)key1 * DHEAD + c1]);
            k0 = *(const v8bf*)&Kp[jo + (size_t)key0 * DHEAD + c0];
            k1 = *(const v8bf*)&Kp[jo + (size_t)key1 * DHEAD + c1];
        }

        // S = Q * K^T  (16 queries x 32 keys, K-dim = 64)
        v8f s0 = {}, s1 = {};
#pragma unroll
        for (int kk = 0; kk < 2; ++kk) {
            v16bf kb0 = load_b_frag(&Kt[buf][(kk * 32) * 40 + 0], 40);
            v16bf kb1 = load_b_frag(&Kt[buf][(kk * 32) * 40 + 16], 40);
            s0 = __builtin_amdgcn_wmma_f32_16x16x32_bf16(false, qa[kk], false, kb0, (short)0, s0, false, false);
            s1 = __builtin_amdgcn_wmma_f32_16x16x32_bf16(false, qa[kk], false, kb1, (short)0, s1, false, false);
        }

        // online softmax in exp2 domain; C layout: row = r + 8*half, key = ln (+16 for s1)
        float f0[8], f1[8];
#pragma unroll
        for (int r = 0; r < 8; ++r) { f0[r] = s0[r] * SCLOG2; f1[r] = s1[r] * SCLOG2; }
#pragma unroll
        for (int r = 0; r < 8; ++r) {
            float t = fmaxf(f0[r], f1[r]);
            t = fmaxf(t, __shfl_xor(t, 1, 32));
            t = fmaxf(t, __shfl_xor(t, 2, 32));
            t = fmaxf(t, __shfl_xor(t, 4, 32));
            t = fmaxf(t, __shfl_xor(t, 8, 32));
            float mn   = fmaxf(m[r], t);
            float corr = exp2f(m[r] - mn);
            float p0 = exp2f(f0[r] - mn);
            float p1 = exp2f(f1[r] - mn);
            float rs = p0 + p1;
            rs += __shfl_xor(rs, 1, 32);
            rs += __shfl_xor(rs, 2, 32);
            rs += __shfl_xor(rs, 4, 32);
            rs += __shfl_xor(rs, 8, 32);
            l[r] = l[r] * corr + rs;
            m[r] = mn;
#pragma unroll
            for (int ni = 0; ni < 4; ++ni) o[ni][r] *= corr;
            int prow = r + 8 * half;
            Ps[wave][prow * 40 + ln]      = f2bf(p0);
            Ps[wave][prow * 40 + 16 + ln] = f2bf(p1);
        }
        asm volatile("s_wait_dscnt 0" ::: "memory");   // P store -> A-frag load (same wave)

        v16bf pa = load_a_frag(&Ps[wave][0], 40);
#pragma unroll
        for (int ni = 0; ni < 4; ++ni) {
            v16bf vb = load_b_frag(&Vs[buf][ni * 16], 72);
            o[ni] = __builtin_amdgcn_wmma_f32_16x16x32_bf16(false, pa, false, vb, (short)0, o[ni], false, false);
        }

        if (more) {                                   // transpose-store prefetched K
#pragma unroll
            for (int j = 0; j < 8; ++j) {
                Kt[buf ^ 1][(c0 + j) * 40 + key0] = k0[j];
                Kt[buf ^ 1][(c1 + j) * 40 + key1] = k1[j];
            }
            asm volatile("s_wait_asynccnt 0x0" ::: "memory");  // V(jt+1) resident
        }
        __syncthreads();
    }

    // normalize and write sa[b,n,h*64+d] as bf16
#pragma unroll
    for (int ni = 0; ni < 4; ++ni)
#pragma unroll
        for (int r = 0; r < 8; ++r) {
            float ov = o[ni][r] / l[r];
            int n   = q0 + wave * 16 + r + 8 * half;
            int col = h * DHEAD + ni * 16 + ln;
            sa[(size_t)(b * SEQ + n) * DIMX + col] = f2bf(ov);
        }
}

extern "C" void kernel_launch(void* const* d_in, const int* in_sizes, int n_in,
                              void* d_out, int out_size, void* d_ws, size_t ws_size,
                              hipStream_t stream) {
    (void)in_sizes; (void)n_in; (void)out_size; (void)ws_size;
    const float* x     = (const float*)d_in[0];
    const float* gamma = (const float*)d_in[1];
    const float* beta  = (const float*)d_in[2];
    const float* w_qkv = (const float*)d_in[3];
    const float* w_out = (const float*)d_in[4];
    const float* b_out = (const float*)d_in[5];
    float* out = (float*)d_out;

    char* ws = (char*)d_ws;
    bf16* wqkv_bf = (bf16*)(ws + 0);          // 1024*3072*2 = 6291456
    bf16* wout_bf = (bf16*)(ws + 6291456);    // 1024*1024*2 = 2097152
    bf16* xn_bf   = (bf16*)(ws + 8388608);    // 4096*1024*2 = 8388608
    bf16* qkv     = (bf16*)(ws + 16777216);   // 3*2*16*2048*64*2 = 25165824
    bf16* sa      = (bf16*)(ws + 41943040);   // 4096*1024*2 = 8388608
    bf16* qb = qkv;
    bf16* kb = qkv + (size_t)BATCH * HEADS * SEQ * DHEAD;
    bf16* vb = qkv + (size_t)2 * BATCH * HEADS * SEQ * DHEAD;

    cvt_bf16_kernel<<<1536, 256, 0, stream>>>(w_qkv, wqkv_bf, DIMX * NQKV);
    cvt_bf16_kernel<<<512,  256, 0, stream>>>(w_out, wout_bf, DIMX * DIMX);
    ln_kernel<<<ROWS, 256, 0, stream>>>(x, gamma, beta, xn_bf);
    gemm_bf16<<<dim3(NQKV / 128, ROWS / 128), 256, 0, stream>>>(
        xn_bf, wqkv_bf, DIMX, NQKV, DIMX, /*mode=*/0, qkv, nullptr, nullptr, NQKV);
    attn_kernel<<<dim3(SEQ / 64, BATCH * HEADS), 128, 0, stream>>>(qb, kb, vb, sa);
    gemm_bf16<<<dim3(DIMX / 128, ROWS / 128), 256, 0, stream>>>(
        sa, wout_bf, DIMX, DIMX, DIMX, /*mode=*/1, nullptr, out, b_out, DIMX);
}
